// SinkhornDistance_16896401343199
// MI455X (gfx1250) — compile-verified, matrix operand order checked by reading, not run on hardware
//
#include <hip/hip_runtime.h>
#include <math.h>

typedef __attribute__((ext_vector_type(16))) _Float16 v16h;
typedef __attribute__((ext_vector_type(8)))  float    v8f;

namespace {
constexpr int    kB      = 8;
constexpr int    kN      = 2048;
constexpr int    kM      = 2048;
constexpr float  kInvEps = 10.0f;     // 1/eps, eps = 0.1
constexpr float  kEps    = 0.1f;
constexpr int    kMaxIter= 50;
constexpr float  kThresh = 1e-12f;
constexpr float  kVClamp = 9e8f;
constexpr float  kLogMu  = -7.6246189861593985f;  // log(1/2048) = log_mu = log_nu
constexpr int    kTS     = 260;       // padded LDS tile row stride (floats); 260%64=4, 16B-aligned rows

// workspace layout in floats
constexpr size_t kUOff    = 0;                              // kB*kN
constexpr size_t kVOff    = kUOff + (size_t)kB * kN;        // kB*kM
constexpr size_t kPmOff   = kVOff + (size_t)kB * kM;        // kB*8*kM  (col partial max)
constexpr size_t kPsOff   = kPmOff + (size_t)kB * 8 * kM;   // kB*8*kM  (col partial sum)
constexpr size_t kErrOff  = kPsOff + (size_t)kB * 8 * kM;   // 1
constexpr size_t kDoneOff = kErrOff + 1;                    // 1 (int)
}

// ---------------------------------------------------------------- init
__global__ __launch_bounds__(256) void sink_init_kernel(float* ws) {
  int i = blockIdx.x * 256 + threadIdx.x;
  if (i < kB * (kN + kM)) ws[i] = 0.0f;           // u = 0, v = 0
  if (i == 0) {
    ws[kErrOff] = 0.0f;
    ((int*)ws)[kDoneOff] = 0;
  }
}

// ---- CDNA5 async global->LDS copy of one 16x256 f32 segment (b128/lane).
// GVS mode: mem_addr = SGPR_base + VGPR_offset; vdst VGPR holds LDS byte addr.
__device__ __forceinline__ void async_load_seg(const float* __restrict__ gbase,
                                               unsigned ldsbase, int seg, int tid) {
#pragma unroll
  for (int i = 0; i < 4; ++i) {
    int idx = tid + i * 256;                 // 0..1023 float4s (16 rows x 64)
    int rr  = idx >> 6;                      // row 0..15
    int k   = idx & 63;                      // float4 within the 256-col row
    unsigned goff  = (unsigned)(rr * (kM * 4) + seg * (256 * 4) + k * 16);
    unsigned laddr = ldsbase + (unsigned)(rr * (kTS * 4) + k * 16);
    asm volatile("global_load_async_to_lds_b128 %0, %1, %2"
                 :: "v"(laddr), "v"(goff), "s"(gbase)
                 : "memory");
  }
}

__device__ __forceinline__ void wait_async0() {
#if __has_builtin(__builtin_amdgcn_s_wait_asynccnt)
  __builtin_amdgcn_s_wait_asynccnt(0);
#else
  asm volatile("s_wait_asynccnt 0x0" ::: "memory");
#endif
}

// ------------------------------------------------- even step: u update
// One block = (batch b, 16-row stripe). 8 waves split each 256-col segment.
// Double-buffered async-to-LDS staging (ASYNCcnt), two passes:
//   pass 1: row max;  pass 2: exp(x-m) -> f16 A tiles, row sums via
//   v_wmma_f32_16x16x32_f16 against an all-ones B matrix.
__global__ __launch_bounds__(256)
void sink_u_kernel(const float* __restrict__ c, float* __restrict__ u,
                   const float* __restrict__ v, float* __restrict__ errAcc,
                   const int* __restrict__ done) {
  if (*done) return;

  __shared__ float tileA[16 * kTS];  // 16.6 KB
  __shared__ float tileB[16 * kTS];  // 16.6 KB
  __shared__ float vsh[kM];          // 8 KB
  __shared__ float ush[16];
  __shared__ float rowmax[8][16];
  __shared__ float partial[8][16];
  __shared__ float du_s[16];

  const int b       = blockIdx.x >> 7;       // 128 row-blocks per batch
  const int rowBase = (blockIdx.x & 127) * 16;
  const int tid     = threadIdx.x;
  const int wave    = tid >> 5;
  const int lane    = tid & 31;
  const int r       = lane & 15;             // row within stripe (WMMA A layout: M = lane%16)
  const int h       = lane >> 4;             // K-half select

  const float*  gbase = c + (size_t)b * kN * kM + (size_t)rowBase * kM;
  const float4* v4    = (const float4*)(v + (size_t)b * kM);

  for (int idx = tid; idx < kM / 4; idx += 256) ((float4*)vsh)[idx] = v4[idx];
  if (tid < 16) ush[tid] = u[(size_t)b * kN + rowBase + tid];

  const unsigned lb0 = (unsigned)(uintptr_t)(void*)tileA;
  const unsigned lb1 = (unsigned)(uintptr_t)(void*)tileB;

  async_load_seg(gbase, lb0, 0, tid);        // prologue: segment 0 -> A
  __syncthreads();                           // vsh/ush visible
  const float ur = ush[r];

  // ---- pass 1: row maxima, 8 double-buffered segments of 256 columns
  float m = -__builtin_inff();
  for (int seg = 0; seg < 8; ++seg) {
    wait_async0();
    __syncthreads();                         // segment `seg` resident, prev buffer free
    if (seg < 7) async_load_seg(gbase, (seg & 1) ? lb0 : lb1, seg + 1, tid);
    const float* tp = (seg & 1) ? tileB : tileA;
    const int base = wave * 32 + h;          // parity-interleaved halves: disjoint banks
    #pragma unroll
    for (int jj = 0; jj < 16; ++jj) {
      int lc = base + 2 * jj;
      float x = (tp[r * kTS + lc] + ur + vsh[seg * 256 + lc]) * kInvEps;
      m = fmaxf(m, x);
    }
  }
  m = fmaxf(m, __shfl_xor(m, 16));           // merge the two K-halves of each row
  if (h == 0) rowmax[wave][r] = m;
  __syncthreads();
  float mrow = rowmax[0][r];
  #pragma unroll
  for (int w = 1; w < 8; ++w) mrow = fmaxf(mrow, rowmax[w][r]);

  // ---- pass 2: exp(x - mrow) -> f16 A tiles, WMMA row-sum vs ones
  v16h ones;
  #pragma unroll
  for (int i = 0; i < 16; ++i) ones[i] = (_Float16)1.0f;
  v8f acc = {};

  async_load_seg(gbase, lb0, 0, tid);        // prologue: segment 0 -> A
  for (int seg = 0; seg < 8; ++seg) {
    wait_async0();
    __syncthreads();
    if (seg < 7) async_load_seg(gbase, (seg & 1) ? lb0 : lb1, seg + 1, tid);
    const float* tp = (seg & 1) ? tileB : tileA;
    v16h a;
    #pragma unroll
    for (int i = 0; i < 16; ++i) {
      // 16-bit A 16x32 layout: lanes<16 hold K={0..7,16..23}, lanes>=16 K={8..15,24..31}
      int k  = h * 8 + i + ((i >= 8) ? 8 : 0);
      int lc = wave * 32 + k;
      float x = (tp[r * kTS + lc] + ur + vsh[seg * 256 + lc]) * kInvEps;
      a[i] = (_Float16)__expf(x - mrow);
    }
    acc = __builtin_amdgcn_wmma_f32_16x16x32_f16(false, a, false, ones,
                                                 (short)0, acc, false, false);
  }

  // C/D layout: VGPR p, lanes 0-15 -> M=p ; lanes 16-31 -> M=8+p (all N cols equal = rowsum)
  if (lane == 0) {
    #pragma unroll
    for (int p = 0; p < 8; ++p) partial[wave][p] = acc[p];
  }
  if (lane == 16) {
    #pragma unroll
    for (int p = 0; p < 8; ++p) partial[wave][8 + p] = acc[p];
  }
  __syncthreads();

  if (tid < 16) {                            // tid==r==lane here (wave 0)
    float s = 0.0f;
    #pragma unroll
    for (int w = 0; w < 8; ++w) s += partial[w][tid];
    float lse  = mrow + logf(s);
    float unew = kEps * (kLogMu - lse) + ur;
    u[(size_t)b * kN + rowBase + tid] = unew;
    du_s[tid] = fabsf(unew - ur);
  }
  __syncthreads();
  if (tid == 0) {
    float e = 0.0f;
    #pragma unroll
    for (int i = 0; i < 16; ++i) e += du_s[i];
    atomicAdd(errAcc, e);
  }
}

__global__ void sink_err_kernel(float* errAcc, int* done) {
  if (!(*done)) {
    float err = (*errAcc) / (float)kB;
    if (err < kThresh) *done = 1;
  }
  *errAcc = 0.0f;
}

// -------------------------------------------------- odd step: v update
// Thread-per-column over a 256-row chunk; fully coalesced; two-pass
// (max then sum-exp) out of L2. Partials (m,s) combined in next kernel.
__global__ __launch_bounds__(256)
void sink_v_pass_kernel(const float* __restrict__ c, const float* __restrict__ u,
                        const float* __restrict__ v, float* __restrict__ pm,
                        float* __restrict__ ps, const int* __restrict__ done) {
  if (*done) return;
  __shared__ float ush[256];

  const int b      = blockIdx.x >> 6;
  const int rem    = blockIdx.x & 63;
  const int colBlk = rem >> 3;
  const int chunk  = rem & 7;
  const int j      = colBlk * 256 + threadIdx.x;
  const int i0     = chunk * 256;

  ush[threadIdx.x] = u[(size_t)b * kN + i0 + threadIdx.x];
  __syncthreads();

  const float vj = v[(size_t)b * kM + j];
  const float* cb = c + (size_t)b * kN * kM + (size_t)i0 * kM + j;

  float m = -__builtin_inff();
  for (int i = 0; i < 256; ++i) {
    if (i + 32 < 256) __builtin_prefetch(cb + (size_t)(i + 32) * kM, 0, 0);
    float x = (cb[(size_t)i * kM] + ush[i] + vj) * kInvEps;
    m = fmaxf(m, x);
  }
  float s = 0.0f;
  for (int i = 0; i < 256; ++i) {
    float x = (cb[(size_t)i * kM] + ush[i] + vj) * kInvEps;
    s += __expf(x - m);
  }
  size_t pidx = ((size_t)b * 8 + chunk) * kM + j;
  pm[pidx] = m;
  ps[pidx] = s;
}

__global__ __launch_bounds__(256)
void sink_v_combine_kernel(float* __restrict__ v, const float* __restrict__ pm,
                           const float* __restrict__ ps, const int* __restrict__ done) {
  if (*done) return;
  const int idx = blockIdx.x * 256 + threadIdx.x;   // 0 .. kB*kM
  const int b = idx >> 11;
  const int j = idx & (kM - 1);

  float m = -__builtin_inff();
  #pragma unroll
  for (int ch = 0; ch < 8; ++ch)
    m = fmaxf(m, pm[((size_t)b * 8 + ch) * kM + j]);
  float s = 0.0f;
  #pragma unroll
  for (int ch = 0; ch < 8; ++ch) {
    size_t p = ((size_t)b * 8 + ch) * kM + j;
    s += ps[p] * __expf(pm[p] - m);
  }
  float lse = m + logf(s);
  float vj  = v[(size_t)b * kM + j];
  float vn  = kEps * (kLogMu - lse) + vj;
  if (vn > kVClamp) vn = 0.0f;                      // torch clamp: zero huge entries
  v[(size_t)b * kM + j] = vn;
}

// -------------------------------------------- final: pi = exp((c+u+v)/eps)
__global__ __launch_bounds__(256)
void sink_pi_kernel(const float* __restrict__ c, const float* __restrict__ u,
                    const float* __restrict__ v, float* __restrict__ out) {
  const size_t t  = (size_t)blockIdx.x * 256 + threadIdx.x;  // 4 elems per thread
  const size_t e0 = t * 4;
  const size_t b  = e0 >> 22;                 // / (2048*2048)
  const size_t rm = e0 & ((1u << 22) - 1);
  const int    i  = (int)(rm >> 11);
  const int    j  = (int)(rm & (kM - 1));

  const float  ui = u[b * kN + i];
  const float4 cc = *(const float4*)(c + e0);
  const float4 vv = *(const float4*)(v + b * kM + j);
  float4 o;
  o.x = __expf((cc.x + ui + vv.x) * kInvEps);
  o.y = __expf((cc.y + ui + vv.y) * kInvEps);
  o.z = __expf((cc.z + ui + vv.z) * kInvEps);
  o.w = __expf((cc.w + ui + vv.w) * kInvEps);
  *(float4*)(out + e0) = o;
}

extern "C" void kernel_launch(void* const* d_in, const int* in_sizes, int n_in,
                              void* d_out, int out_size, void* d_ws, size_t ws_size,
                              hipStream_t stream) {
  const float* c  = (const float*)d_in[0];
  float*       ws = (float*)d_ws;

  float* u      = ws + kUOff;
  float* v      = ws + kVOff;
  float* pm     = ws + kPmOff;
  float* ps     = ws + kPsOff;
  float* errAcc = ws + kErrOff;
  int*   done   = (int*)(ws + kDoneOff);

  sink_init_kernel<<<(kB * (kN + kM) + 255) / 256, 256, 0, stream>>>(ws);

  for (int it = 0; it < kMaxIter; ++it) {
    if ((it & 1) == 0) {
      sink_u_kernel<<<kB * (kN / 16), 256, 0, stream>>>(c, u, v, errAcc, done);
      sink_err_kernel<<<1, 1, 0, stream>>>(errAcc, done);
    } else {
      sink_v_pass_kernel<<<kB * 8 * 8, 256, 0, stream>>>(c, u, v, pm, ps, done);
      sink_v_combine_kernel<<<(kB * kM) / 256, 256, 0, stream>>>(v, pm, ps, done);
    }
  }

  sink_pi_kernel<<<(int)(((size_t)kB * kN * kM / 4) / 256), 256, 0, stream>>>(c, u, v, (float*)d_out);
}
